// KnowledgeGuidedImputation_71270687309840
// MI455X (gfx1250) — compile-verified, matrix operand order checked by reading, not run on hardware
//
#include <hip/hip_runtime.h>
#include <hip/hip_bf16.h>
#include <stdint.h>

// ---------------------------------------------------------------------------
// KnowledgeGuidedImputation for MI455X (gfx1250, wave32, WMMA bf16)
//
// All GEMM B-operands are prepacked into fragment-native layout
// [K/32][N][32] so every WMMA operand load is 2 x b128 (ISA 7.12.2: a lane's
// 16 bf16 fragment elements are two contiguous 8-element K-runs).
// GEMMs are LDS-free / barrier-free: one 16x16 output tile per wave.
// ---------------------------------------------------------------------------

typedef __bf16 bf16_t;
typedef __bf16 v16bf __attribute__((ext_vector_type(16)));
typedef float  v8f   __attribute__((ext_vector_type(8)));

#define B_  32
#define T_  256
#define F_  64
#define D_  768
#define S_  768
#define H_  256
#define NH_ 4
#define HD_ 64
#define BT_ (B_*T_)   // 8192
#define P_  (F_*F_)   // 4096

union Frag {
    uint4 u4[2];
    v16bf v;
};

// element e of an A/B fragment maps to K = frag_k(e, half) (ISA 7.12.2)
__device__ __forceinline__ int frag_k(int e, int half) {
    return (((e & 8) << 1) | (e & 7)) + (half << 3);
}

// load a fragment whose 16 elements live at base[half*8 + {0..7}] and
// base[16 + half*8 + {0..7}] (K-contiguous storage)
__device__ __forceinline__ v16bf load_frag(const bf16_t* base, int half) {
    Frag f;
    const bf16_t* p = base + (half << 3);
    f.u4[0] = *reinterpret_cast<const uint4*>(p);
    f.u4[1] = *reinterpret_cast<const uint4*>(p + 16);
    return f.v;
}

// ---------------------------------------------------------------------------
// Small prep kernels
// ---------------------------------------------------------------------------
__global__ void present_mask_ker(const uint8_t* __restrict__ pp,
                                 unsigned long long* __restrict__ pm) {
    int i = threadIdx.x;
    if (i < F_) {
        unsigned long long m = 0ull;
        for (int j = i + 1; j < F_; ++j)
            if (pp[i * F_ + j]) m |= (1ull << j);
        pm[i] = m;
    }
}

__global__ void prep_ker(const float* __restrict__ batch,
                         const float* __restrict__ probs,
                         const unsigned long long* __restrict__ pmG,
                         bf16_t* __restrict__ safe,
                         unsigned long long* __restrict__ surv,
                         float* __restrict__ cnt,
                         float* __restrict__ stochOut) {
    __shared__ unsigned long long pm[F_];
    if (threadIdx.x < F_) pm[threadIdx.x] = pmG[threadIdx.x];
    __syncthreads();
    int row = blockIdx.x * blockDim.x + threadIdx.x;   // 0..8191
    unsigned long long m = 0ull;
    for (int f = 0; f < F_; ++f) {
        float x = batch[(size_t)row * F_ + f];
        bool valid = (x == x);                           // !isnan
        bool st = (probs[(size_t)row * F_ + f] < 0.3f) && valid;
        bool sv = valid && !st;
        safe[(size_t)row * F_ + f] = sv ? (bf16_t)x : (bf16_t)0.0f;
        stochOut[(size_t)row * F_ + f] = st ? 1.0f : 0.0f;
        if (sv) m |= (1ull << f);
    }
    surv[row] = m;
    float c = 0.0f;
    for (int i = 0; i < F_; ++i)
        if ((m >> i) & 1ull) c += (float)__popcll(m & pm[i]);
    cnt[row] = c;
}

// pack f32 W[K,N] -> bf16 Wpk[K/32][N][32] (fragment-native), output-indexed
__global__ void pack_w_ker(const float* __restrict__ W,
                           bf16_t* __restrict__ out, int N) {
    int idx = blockIdx.x * blockDim.x + threadIdx.x;    // < K*N
    int kt  = idx / (N * 32);
    int rem = idx - kt * (N * 32);
    int n   = rem >> 5, kl = rem & 31;
    int k   = kt * 32 + kl;
    out[idx] = (bf16_t)W[(size_t)k * N + n];
}

// pack pair_emb[4096,768] with (i<j && present) row zeroing, fragment-native
__global__ void bpair_pack_ker(const float* __restrict__ pair_emb,
                               const uint8_t* __restrict__ pp,
                               bf16_t* __restrict__ out) {
    int idx = blockIdx.x * blockDim.x + threadIdx.x;    // < 4096*768
    int kt  = idx / (D_ * 32);
    int rem = idx - kt * (D_ * 32);
    int d   = rem >> 5, kl = rem & 31;
    int p   = kt * 32 + kl;
    int i = p >> 6, j = p & 63;
    bool keep = (i < j) && (pp[p] != 0);
    out[idx] = keep ? (bf16_t)pair_emb[(size_t)p * D_ + d] : (bf16_t)0.0f;
}

__global__ void means_ker(const float* __restrict__ sap,
                          float* __restrict__ meanS) {
    int s = blockIdx.x * blockDim.x + threadIdx.x;
    if (s < S_) {
        float a = 0.0f;
        for (int f = 0; f < F_; ++f) a += sap[(size_t)f * S_ + s];
        meanS[s] = a * (1.0f / (float)F_);
    }
}

__global__ void qbias_ker(const float* __restrict__ meanS,
                          const float* __restrict__ W_sap,
                          const float* __restrict__ b_sap,
                          const float* __restrict__ b_num,
                          float* __restrict__ qbias) {
    int h = threadIdx.x;
    if (h < H_) {
        float a = b_num[h] + b_sap[h];
        for (int s = 0; s < S_; ++s) a += meanS[s] * W_sap[(size_t)s * H_ + h];
        qbias[h] = a;
    }
}

__global__ void ctxnorm_ker(const float* __restrict__ sumemb,
                            const float* __restrict__ cnt,
                            bf16_t* __restrict__ ctxb) {
    int idx = blockIdx.x * blockDim.x + threadIdx.x;   // < 8192*768
    int row = idx / D_;
    float c = cnt[row];
    float v = (c > 0.0f) ? sumemb[idx] / c : 0.0f;
    ctxb[idx] = (bf16_t)v;
}

__global__ void attnmean_ker(const float* __restrict__ attnW,
                             float* __restrict__ outAttn) {
    int idx = blockIdx.x * blockDim.x + threadIdx.x;   // < 32*256*256
    int b = idx / (T_ * T_);
    int rest = idx - b * (T_ * T_);
    size_t base = (size_t)b * NH_ * T_ * T_;
    float s = 0.0f;
#pragma unroll
    for (int h = 0; h < NH_; ++h)
        s += attnW[base + (size_t)h * T_ * T_ + rest];
    outAttn[idx] = s * 0.25f;
}

// ---------------------------------------------------------------------------
// Direct WMMA GEMM (no LDS, no barriers): one 16x16 tile per wave.
// A[M,K] row-major bf16 (K%32==0), Bpk packed [K/32][N][32].
// C = A@B (+bias) (+addend); outputs: Cf (f32), Cb (bf16), CbT (bf16, N-major)
// ---------------------------------------------------------------------------
__global__ __launch_bounds__(256)
void gemm_direct_ker(const bf16_t* __restrict__ A, const bf16_t* __restrict__ Bpk,
                     const float* __restrict__ bias, const float* __restrict__ addend,
                     float* __restrict__ Cf, bf16_t* __restrict__ Cb,
                     bf16_t* __restrict__ CbT,
                     int M, int N, int K, int nt16 /* = N/16 */) {
    const int tid  = threadIdx.x;
    const int lane = tid & 31, wv = tid >> 5;
    const int lm   = lane & 15, half = lane >> 4;
    const int tile = blockIdx.x * 8 + wv;
    const int tm   = tile / nt16, tn = tile - tm * nt16;
    const int m0   = tm * 16,     n0 = tn * 16;
    const int col  = n0 + lm;

    v8f acc;
#pragma unroll
    for (int r = 0; r < 8; ++r) acc[r] = 0.0f;

    const bf16_t* ap = A + (size_t)(m0 + lm) * K;              // row lm of A tile
    const bf16_t* bp = Bpk + (size_t)col * 32;                 // kt = 0 slice
    const size_t  bstep = (size_t)N * 32;

    for (int k0 = 0; k0 < K; k0 += 32) {
        v16bf a = load_frag(ap + k0, half);
        v16bf b = load_frag(bp, half);
        __builtin_prefetch(bp + bstep, 0, 1);                  // global_prefetch
        bp += bstep;
        acc = __builtin_amdgcn_wmma_f32_16x16x32_bf16(false, a, false, b,
                                                      (short)0, acc, false, false);
    }
#pragma unroll
    for (int r = 0; r < 8; ++r) {
        int row = m0 + (half << 3) + r;
        float v = acc[r];
        if (bias)   v += bias[col];
        if (addend) v += addend[(size_t)row * N + col];
        if (Cf)  Cf [(size_t)row * N + col] = v;
        if (Cb)  Cb [(size_t)row * N + col] = (bf16_t)v;
        if (CbT) CbT[(size_t)col * M + row] = (bf16_t)v;
    }
}

// ---------------------------------------------------------------------------
// Pair GEMM: sum_emb[8192,768] = A(mask bits) @ Bpair_pk
// A[bt, p=(i,j)] = surv_bit_i & surv_bit_j, synthesized in VGPRs (no loads).
// ---------------------------------------------------------------------------
__global__ __launch_bounds__(256)
void pair_gemm_ker(const unsigned long long* __restrict__ surv,
                   const bf16_t* __restrict__ Bpk,
                   float* __restrict__ sumemb) {
    const int tid  = threadIdx.x;
    const int lane = tid & 31, wv = tid >> 5;
    const int lm   = lane & 15, half = lane >> 4;
    const int tile = blockIdx.x * 8 + wv;
    const int tm   = tile / (D_ / 16), tn = tile - tm * (D_ / 16);
    const int m0   = tm * 16, n0 = tn * 16;
    const int col  = n0 + lm;

    const unsigned long long mlane = surv[m0 + lm];
    const bf16_t one = (bf16_t)1.0f, zero = (bf16_t)0.0f;

    v8f acc;
#pragma unroll
    for (int r = 0; r < 8; ++r) acc[r] = 0.0f;

    const bf16_t* bp = Bpk + (size_t)col * 32;
    const size_t  bstep = (size_t)D_ * 32;

    for (int k0 = 0; k0 < P_; k0 += 32) {
        v16bf a;
#pragma unroll
        for (int e = 0; e < 16; ++e) {
            int kl = frag_k(e, half);
            int p = k0 + kl;
            int i = p >> 6, j = p & 63;
            unsigned long long bit = (mlane >> i) & (mlane >> j) & 1ull;
            a[e] = bit ? one : zero;
        }
        v16bf b = load_frag(bp, half);
        __builtin_prefetch(bp + bstep, 0, 1);
        bp += bstep;
        acc = __builtin_amdgcn_wmma_f32_16x16x32_bf16(false, a, false, b,
                                                      (short)0, acc, false, false);
    }
#pragma unroll
    for (int r = 0; r < 8; ++r) {
        int row = m0 + (half << 3) + r;
        sumemb[(size_t)row * D_ + col] = acc[r];
    }
}

// ---------------------------------------------------------------------------
// Attention: per (b, head, 16-query tile): scores WMMA -> softmax -> attn@V
// blockDim = 128 (4 waves). K tile staged row-major [256,64] (K-contiguous
// per key => b128 fragments). V staged d-major [64,256] from transposed vT
// (K-contiguous per out-dim => b128 fragments).
// ---------------------------------------------------------------------------
__global__ __launch_bounds__(128)
void attn_ker(const bf16_t* __restrict__ qb, const bf16_t* __restrict__ kb,
              const bf16_t* __restrict__ vT,
              float* __restrict__ attnW, bf16_t* __restrict__ attnout) {
    __shared__ bf16_t ldsQ[16 * HD_];      // 2 KB  [query][d]
    __shared__ bf16_t ldsKV[T_ * HD_];     // 32 KB [key][d], then [d][key]
    __shared__ float  ldsS[16 * T_];       // 16 KB scores/exp
    __shared__ bf16_t ldsP[16 * T_];       // 8 KB probs [query][key]
    __shared__ float  ldsRed[16 * 8];
    __shared__ float  ldsMax[16];
    __shared__ float  ldsSum[16];

    const int tid  = threadIdx.x;
    const int lane = tid & 31, wv = tid >> 5;
    const int lm   = lane & 15, half = lane >> 4;
    const int z = blockIdx.y;              // b*NH + h
    const int b = z >> 2, h = z & 3;
    const int t0 = blockIdx.x * 16;

    // ---- stage Q [16,64] and K [256,64], both row-major ----
    {
        int idx = tid * 8;                 // 1024 elems / 128 threads
        int r = idx >> 6, c = idx & 63;
        *reinterpret_cast<uint4*>(&ldsQ[idx]) =
            *reinterpret_cast<const uint4*>(&qb[((size_t)(b * T_ + t0 + r)) * H_ + h * HD_ + c]);
    }
    for (int idx = tid * 8; idx < T_ * HD_; idx += 128 * 8) {
        int r = idx >> 6, c = idx & 63;
        *reinterpret_cast<uint4*>(&ldsKV[idx]) =
            *reinterpret_cast<const uint4*>(&kb[((size_t)(b * T_ + r)) * H_ + h * HD_ + c]);
    }
    __syncthreads();

    // ---- scores: 16 x 256; wave wv covers keys [wv*64, wv*64+64) ----
    v8f sacc[4];
#pragma unroll
    for (int t = 0; t < 4; ++t)
#pragma unroll
        for (int r = 0; r < 8; ++r) sacc[t][r] = 0.0f;

#pragma unroll
    for (int k0 = 0; k0 < HD_; k0 += 32) {
        v16bf a = load_frag(&ldsQ[lm * HD_ + k0], half);
#pragma unroll
        for (int t = 0; t < 4; ++t) {
            // B[k=d][n=key] = K[key][d]: key row is K-contiguous
            v16bf bf = load_frag(&ldsKV[(wv * 64 + t * 16 + lm) * HD_ + k0], half);
            sacc[t] = __builtin_amdgcn_wmma_f32_16x16x32_bf16(false, a, false, bf,
                                                              (short)0, sacc[t], false, false);
        }
    }
#pragma unroll
    for (int t = 0; t < 4; ++t)
#pragma unroll
        for (int r = 0; r < 8; ++r)
            ldsS[((half << 3) + r) * T_ + wv * 64 + t * 16 + lm] = sacc[t][r] * 0.125f;
    __syncthreads();

    // ---- softmax over 256 keys per row; 8 threads per row ----
    {
        int row = tid >> 3, part = tid & 7, base = part * 32;
        float mx = -3.402823466e+38f;
        for (int c = 0; c < 32; ++c) mx = fmaxf(mx, ldsS[row * T_ + base + c]);
        ldsRed[row * 8 + part] = mx;
        __syncthreads();
        if (part == 0) {
            float m = ldsRed[row * 8];
            for (int p = 1; p < 8; ++p) m = fmaxf(m, ldsRed[row * 8 + p]);
            ldsMax[row] = m;
        }
        __syncthreads();
        float rm = ldsMax[row], sum = 0.0f;
        for (int c = 0; c < 32; ++c) {
            float e = __expf(ldsS[row * T_ + base + c] - rm);
            ldsS[row * T_ + base + c] = e;
            sum += e;
        }
        ldsRed[row * 8 + part] = sum;
        __syncthreads();
        if (part == 0) {
            float s = 0.0f;
            for (int p = 0; p < 8; ++p) s += ldsRed[row * 8 + p];
            ldsSum[row] = s;
        }
        __syncthreads();
        float inv = 1.0f / ldsSum[row];
        for (int c = 0; c < 32; ++c) {
            float pr = ldsS[row * T_ + base + c] * inv;
            attnW[((size_t)z * T_ + t0 + row) * T_ + base + c] = pr;
            ldsP[row * T_ + base + c] = (bf16_t)pr;
        }
    }
    __syncthreads();

    // ---- reload ldsKV with V, d-major [64,256] from vT[col][row] ----
    for (int idx = tid * 8; idx < T_ * HD_; idx += 128 * 8) {
        int d = idx >> 8, key = idx & 255;
        *reinterpret_cast<uint4*>(&ldsKV[idx]) =
            *reinterpret_cast<const uint4*>(&vT[((size_t)(h * HD_ + d)) * BT_ + b * T_ + key]);
    }
    __syncthreads();

    // ---- attn @ V : [16,256] @ [256,64]; wave wv covers out cols wv*16.. ----
    v8f oacc;
#pragma unroll
    for (int r = 0; r < 8; ++r) oacc[r] = 0.0f;
#pragma unroll
    for (int k0 = 0; k0 < T_; k0 += 32) {
        v16bf a  = load_frag(&ldsP[lm * T_ + k0], half);
        // B[k=key][n=d] = V[key][d] = ldsKV[d][key]: d row is key-contiguous
        v16bf bf = load_frag(&ldsKV[(wv * 16 + lm) * T_ + k0], half);
        oacc = __builtin_amdgcn_wmma_f32_16x16x32_bf16(false, a, false, bf,
                                                       (short)0, oacc, false, false);
    }
#pragma unroll
    for (int r = 0; r < 8; ++r) {
        int row = t0 + (half << 3) + r;
        int colh = h * HD_ + wv * 16 + lm;
        attnout[((size_t)(b * T_ + row)) * H_ + colh] = (bf16_t)oacc[r];
    }
}

// ---------------------------------------------------------------------------
// Host launcher
// ---------------------------------------------------------------------------
extern "C" void kernel_launch(void* const* d_in, const int* in_sizes, int n_in,
                              void* d_out, int out_size, void* d_ws, size_t ws_size,
                              hipStream_t stream) {
    const float*   batch  = (const float*)d_in[0];
    const float*   probs  = (const float*)d_in[1];
    const float*   sap    = (const float*)d_in[2];
    const float*   pemb   = (const float*)d_in[3];
    const uint8_t* ppres  = (const uint8_t*)d_in[4];
    const float *W_num = (const float*)d_in[5],  *b_num = (const float*)d_in[6];
    const float *W_sap = (const float*)d_in[7],  *b_sap = (const float*)d_in[8];
    const float *W_ad  = (const float*)d_in[9],  *b_ad  = (const float*)d_in[10];
    const float *Wq = (const float*)d_in[11], *bq = (const float*)d_in[12];
    const float *Wk = (const float*)d_in[13], *bk = (const float*)d_in[14];
    const float *Wv = (const float*)d_in[15], *bv = (const float*)d_in[16];
    const float *Wo = (const float*)d_in[17], *bo = (const float*)d_in[18];
    const float *W_out = (const float*)d_in[19], *b_out = (const float*)d_in[20];

    float* out = (float*)d_out;
    float* outImputed = out;                          // [8192, 64]
    float* outStoch   = out + (size_t)BT_ * F_;       // [8192, 64]
    float* outAttn    = out + (size_t)2 * BT_ * F_;   // [32, 256, 256]

    char* ws = (char*)d_ws;
    size_t off = 0;
    auto alloc = [&](size_t bytes) -> void* {
        void* p = ws + off;
        off = (off + bytes + 255) & ~(size_t)255;
        return p;
    };
    unsigned long long* survM = (unsigned long long*)alloc(BT_ * 8);
    float*   cnt     = (float*)alloc(BT_ * 4);
    unsigned long long* pmask = (unsigned long long*)alloc(F_ * 8);
    bf16_t*  safeB   = (bf16_t*)alloc((size_t)BT_ * F_ * 2);
    bf16_t*  BpairP  = (bf16_t*)alloc((size_t)P_ * D_ * 2);
    float*   sumemb  = (float*)alloc((size_t)BT_ * D_ * 4);
    bf16_t*  ctxB    = (bf16_t*)alloc((size_t)BT_ * D_ * 2);
    float*   queryF  = (float*)alloc((size_t)BT_ * H_ * 4);
    bf16_t*  queryB  = (bf16_t*)alloc((size_t)BT_ * H_ * 2);
    bf16_t*  kvB     = (bf16_t*)alloc((size_t)BT_ * H_ * 2);
    bf16_t*  qB      = (bf16_t*)alloc((size_t)BT_ * H_ * 2);
    bf16_t*  kB      = (bf16_t*)alloc((size_t)BT_ * H_ * 2);
    bf16_t*  vT      = (bf16_t*)alloc((size_t)BT_ * H_ * 2);   // [H, BT]
    bf16_t*  aoB     = (bf16_t*)alloc((size_t)BT_ * H_ * 2);
    bf16_t*  fusedB  = (bf16_t*)alloc((size_t)BT_ * H_ * 2);
    float*   attnW   = (float*)alloc((size_t)B_ * NH_ * T_ * T_ * 4);
    float*   meanS   = (float*)alloc(S_ * 4);
    float*   qbias   = (float*)alloc(H_ * 4);
    bf16_t*  WnumP   = (bf16_t*)alloc((size_t)F_ * H_ * 2);
    bf16_t*  WadP    = (bf16_t*)alloc((size_t)D_ * H_ * 2);
    bf16_t*  WqP     = (bf16_t*)alloc((size_t)H_ * H_ * 2);
    bf16_t*  WkP     = (bf16_t*)alloc((size_t)H_ * H_ * 2);
    bf16_t*  WvP     = (bf16_t*)alloc((size_t)H_ * H_ * 2);
    bf16_t*  WoP     = (bf16_t*)alloc((size_t)H_ * H_ * 2);
    bf16_t*  WoutP   = (bf16_t*)alloc((size_t)H_ * F_ * 2);
    (void)ws_size; (void)n_in; (void)in_sizes; (void)out_size;

    // masks, safe data, counts, stoch output
    present_mask_ker<<<1, 64, 0, stream>>>(ppres, pmask);
    prep_ker<<<BT_ / 256, 256, 0, stream>>>(batch, probs, pmask, safeB, survM, cnt, outStoch);

    // weight prepack (f32 -> bf16, fragment-native [K/32][N][32])
    pack_w_ker<<<(F_ * H_) / 256, 256, 0, stream>>>(W_num, WnumP, H_);
    pack_w_ker<<<(D_ * H_) / 256, 256, 0, stream>>>(W_ad,  WadP,  H_);
    pack_w_ker<<<(H_ * H_) / 256, 256, 0, stream>>>(Wq, WqP, H_);
    pack_w_ker<<<(H_ * H_) / 256, 256, 0, stream>>>(Wk, WkP, H_);
    pack_w_ker<<<(H_ * H_) / 256, 256, 0, stream>>>(Wv, WvP, H_);
    pack_w_ker<<<(H_ * H_) / 256, 256, 0, stream>>>(Wo, WoP, H_);
    pack_w_ker<<<(H_ * F_) / 256, 256, 0, stream>>>(W_out, WoutP, F_);

    // pooled sapbert bias folded into h_num bias
    means_ker<<<S_ / 256, 256, 0, stream>>>(sap, meanS);
    qbias_ker<<<1, 256, 0, stream>>>(meanS, W_sap, b_sap, b_num, qbias);

    // pair-embedding pack + the big GEMM + normalization
    bpair_pack_ker<<<(P_ * D_) / 256, 256, 0, stream>>>(pemb, ppres, BpairP);
    pair_gemm_ker<<<(BT_ / 16) * (D_ / 16) / 8, 256, 0, stream>>>(survM, BpairP, sumemb);
    ctxnorm_ker<<<(BT_ * D_) / 256, 256, 0, stream>>>(sumemb, cnt, ctxB);

    // query = safe @ W_num + (b_num + pooled)  (f32 + bf16 outputs)
    gemm_direct_ker<<<(BT_ / 16) * (H_ / 16) / 8, 256, 0, stream>>>(
        safeB, WnumP, qbias, nullptr, queryF, queryB, nullptr, BT_, H_, F_, H_ / 16);
    // kv = ctx @ W_ad + b_ad
    gemm_direct_ker<<<(BT_ / 16) * (H_ / 16) / 8, 256, 0, stream>>>(
        ctxB, WadP, b_ad, nullptr, nullptr, kvB, nullptr, BT_, H_, D_, H_ / 16);
    // q/k/v projections (v written transposed for the attn@V fragment path)
    gemm_direct_ker<<<(BT_ / 16) * (H_ / 16) / 8, 256, 0, stream>>>(
        queryB, WqP, bq, nullptr, nullptr, qB, nullptr, BT_, H_, H_, H_ / 16);
    gemm_direct_ker<<<(BT_ / 16) * (H_ / 16) / 8, 256, 0, stream>>>(
        kvB, WkP, bk, nullptr, nullptr, kB, nullptr, BT_, H_, H_, H_ / 16);
    gemm_direct_ker<<<(BT_ / 16) * (H_ / 16) / 8, 256, 0, stream>>>(
        kvB, WvP, bv, nullptr, nullptr, nullptr, vT, BT_, H_, H_, H_ / 16);

    // attention
    attn_ker<<<dim3(T_ / 16, B_ * NH_), 128, 0, stream>>>(qB, kB, vT, attnW, aoB);

    // fused = query + attnout @ Wo + bo
    gemm_direct_ker<<<(BT_ / 16) * (H_ / 16) / 8, 256, 0, stream>>>(
        aoB, WoP, bo, queryF, nullptr, fusedB, nullptr, BT_, H_, H_, H_ / 16);
    // imputed = fused @ W_out + b_out -> d_out
    gemm_direct_ker<<<(BT_ / 16) * (F_ / 16) / 8, 256, 0, stream>>>(
        fusedB, WoutP, b_out, nullptr, outImputed, nullptr, nullptr, BT_, F_, H_, F_ / 16);

    // attn weights averaged over heads
    attnmean_ker<<<(B_ * T_ * T_) / 256, 256, 0, stream>>>(attnW, outAttn);
}
